// GraphAttentionBase_44040594653920
// MI455X (gfx1250) — matrix operand order repaired and checked by reading, not compiled, not run on hardware
//
#include <hip/hip_runtime.h>
#include <hip/hip_bf16.h>
#include <math.h>

#define NN 8192
#define FF 512
#define UU 128
#define NW (NN / 32)      // 256 mask words per row
#define DCHUNKS 16        // partial-sum chunks for denominator
#define MSTRIDE 260       // padded LDS mask row stride (words): 16B-aligned, bank-rotating

typedef __attribute__((ext_vector_type(16))) __bf16 v16bf;
typedef __attribute__((ext_vector_type(8)))  __bf16 v8bf;
typedef __attribute__((ext_vector_type(8)))  float  v8f;

__device__ __forceinline__ float lrelu(float x) { return x > 0.f ? x : 0.2f * x; }

__device__ __forceinline__ v16bf join16(v8bf lo, v8bf hi) {
  v16bf r;
#pragma unroll
  for (int e = 0; e < 8; ++e) { r[e] = lo[e]; r[8 + e] = hi[e]; }
  return r;
}

// ---------------------------------------------------------------- prep kernels

__global__ void k_cvt_kernel(const float* __restrict__ kern, __bf16* __restrict__ kbf) {
  int idx = blockIdx.x * blockDim.x + threadIdx.x;       // UU*FF threads
  kbf[idx] = (__bf16)kern[idx];                          // kernel is [UU][FF]: row u = K-contig
}

__global__ void k_cvt_dw(const float* __restrict__ dw, __bf16* __restrict__ dwT) {
  int idx = blockIdx.x * blockDim.x + threadIdx.x;       // UU*UU threads
  int n = idx / UU, k = idx % UU;
  dwT[idx] = (__bf16)dw[k * UU + n];                     // dwT[n][k] = dense_w[k][n]
}

__global__ void k_init_smax(float* __restrict__ smax) {
  int j = blockIdx.x * blockDim.x + threadIdx.x;
  smax[j] = __int_as_float(0xFF800000);                  // -inf
}

// -------------------------------------------------- ft = features @ kernel^T (WMMA)

__global__ void k_ft(const float* __restrict__ feat, const __bf16* __restrict__ kbf,
                     float* __restrict__ ft) {
  const int lane = threadIdx.x & 31;
  const int wave = threadIdx.x >> 5;
  const int i0 = blockIdx.x * 16;
  const int r = lane & 15;
  const int hi = lane >> 4;
  const int u0a = (wave * 2 + 0) * 16;
  const int u0b = (wave * 2 + 1) * 16;
  v8f acc0 = {}, acc1 = {};
  for (int k0 = 0; k0 < FF; k0 += 32) {
    // A-frag: row i0+r; lanes 0-15: K 0..15; lanes 16-31: K 16..31 (contig)
    const int s0 = hi ? 16 : 0;
    const float* arow = feat + (size_t)(i0 + r) * FF + k0 + s0;
    float4 a0 = *(const float4*)(arow);
    float4 a1 = *(const float4*)(arow + 4);
    float4 a2 = *(const float4*)(arow + 8);
    float4 a3 = *(const float4*)(arow + 12);
    float av[16] = {a0.x, a0.y, a0.z, a0.w, a1.x, a1.y, a1.z, a1.w,
                    a2.x, a2.y, a2.z, a2.w, a3.x, a3.y, a3.z, a3.w};
    v16bf Af;
#pragma unroll
    for (int e = 0; e < 16; ++e) Af[e] = (__bf16)av[e];
    // B-frags: col u = u0+r; lanes 0-15 K=k0..k0+15, lanes 16-31 K=k0+16..k0+31 (contig)
    const int kb = k0 + (hi ? 16 : 0);
    const __bf16* b0p = kbf + (size_t)(u0a + r) * FF + kb;
    const __bf16* b1p = kbf + (size_t)(u0b + r) * FF + kb;
    v16bf B0 = join16(*(const v8bf*)b0p, *(const v8bf*)(b0p + 8));
    v16bf B1 = join16(*(const v8bf*)b1p, *(const v8bf*)(b1p + 8));
    acc0 = __builtin_amdgcn_wmma_f32_16x16x32_bf16(false, Af, false, B0, (short)0, acc0, false, false);
    acc1 = __builtin_amdgcn_wmma_f32_16x16x32_bf16(false, Af, false, B1, (short)0, acc1, false, false);
  }
#pragma unroll
  for (int rr = 0; rr < 8; ++rr) {
    const int i = i0 + rr + (hi ? 8 : 0);
    ft[(size_t)i * UU + u0a + r] = acc0[rr];
    ft[(size_t)i * UU + u0b + r] = acc1[rr];
  }
}

// ------------------------------------------------ transpose ft -> ftT (bf16, [u][i])

__global__ void k_tr(const float* __restrict__ ft, __bf16* __restrict__ ftT) {
  __shared__ float tile[32][33];
  const int tx = threadIdx.x, ty = threadIdx.y;          // block (32,8)
  const int u0 = blockIdx.x * 32, i0 = blockIdx.y * 32;
#pragma unroll
  for (int k = 0; k < 32; k += 8)
    tile[ty + k][tx] = ft[(size_t)(i0 + ty + k) * UU + u0 + tx];
  __syncthreads();
#pragma unroll
  for (int k = 0; k < 32; k += 8)
    ftT[(size_t)(u0 + ty + k) * NN + i0 + tx] = (__bf16)tile[tx][ty + k];
}

// ------------------------------------------------------- s = ft@a_self, c = ft@a_cross

__global__ void k_sc(const float* __restrict__ ft, const float* __restrict__ aself,
                     const float* __restrict__ across, float* __restrict__ s,
                     float* __restrict__ c) {
  __shared__ float rs[UU], rc[UU];
  const int row = blockIdx.x, t = threadIdx.x;
  const float v = ft[(size_t)row * UU + t];
  rs[t] = v * aself[t];
  rc[t] = v * across[t];
  __syncthreads();
  for (int st = 64; st > 0; st >>= 1) {
    if (t < st) { rs[t] += rs[t + st]; rc[t] += rc[t + st]; }
    __syncthreads();
  }
  if (t == 0) { s[row] = rs[0]; c[row] = rc[0]; }
}

// ------------------- pass 1a: bitmask of A + per-column sMax (single read of 256MB A)

__global__ void k_mask(const float* __restrict__ A, const float* __restrict__ s,
                       unsigned* __restrict__ mask, float* __restrict__ smax) {
  const int lane = threadIdx.x & 31;
  const int wave = threadIdx.x >> 5;
  const int g = blockIdx.x;                              // column group 0..255
  const int j = g * 32 + lane;
  const int chunk = blockIdx.y * 8 + wave;               // 0..255
  const int iStart = chunk * (NN / 256);                 // 32 rows each
  float sm = -INFINITY;
  for (int k = 0; k < NN / 256; ++k) {
    const int i = iStart + k;
    const float a = A[(size_t)i * NN + j];
    const bool e = (a != 0.f);
    const unsigned word = __builtin_amdgcn_ballot_w32(e);
    if (lane == 0) mask[(size_t)i * NW + g] = word;
    const float sv = s[i];
    sm = fmaxf(sm, e ? sv : -INFINITY);
  }
  if (sm != -INFINITY) {                                 // order-independent float atomic max
    if (sm >= 0.f) atomicMax((int*)&smax[j], __float_as_int(sm));
    else           atomicMin((unsigned int*)&smax[j], (unsigned int)__float_as_int(sm));
  }
}

// ----------------- pass 1b: column max m[j] = lrelu(sMax[j] + c[j]) (monotonic lrelu)

__global__ void k_colmax(const float* __restrict__ smax, const float* __restrict__ c,
                         float* __restrict__ m) {
  const int j = blockIdx.x * blockDim.x + threadIdx.x;
  m[j] = lrelu(smax[j] + c[j]);
}

// ----------------- pass 1c: denominator partial sums over row-chunks (deterministic)

__global__ void k_dpart(const unsigned* __restrict__ mask, const float* __restrict__ s,
                        const float* __restrict__ c, const float* __restrict__ m,
                        float* __restrict__ dPart) {
  const int lane = threadIdx.x & 31;
  const int wave = threadIdx.x >> 5;
  const int g = blockIdx.x;
  const int j = g * 32 + lane;
  const int chunk = blockIdx.y * 8 + wave;               // 0..DCHUNKS-1
  const int rows = NN / DCHUNKS;
  const int iStart = chunk * rows;
  const float cj = c[j], mj = m[j];
  float acc = 0.f;
  for (int k = 0; k < rows; ++k) {
    const int i = iStart + k;
    const unsigned word = mask[(size_t)i * NW + g];      // lane-uniform -> scalar load/branch
    if (word) {
      if ((word >> lane) & 1u) acc += __expf(lrelu(s[i] + cj) - mj);
    }
  }
  dPart[(size_t)chunk * NN + j] = acc;
}

__global__ void k_dred(const float* __restrict__ dPart, float* __restrict__ rd) {
  const int j = blockIdx.x * blockDim.x + threadIdx.x;
  float d = 0.f;
#pragma unroll
  for (int p = 0; p < DCHUNKS; ++p) d += dPart[(size_t)p * NN + j];
  rd[j] = d > 0.f ? 1.f / d : 0.f;
}

// --------------------- pass 2: out_mid = P @ ft  (P built in LDS per 16x32 K-chunk)
// Block's 16 mask rows (16KB) are staged into LDS once via async-to-LDS DMA.

__global__ void k_att(const unsigned* __restrict__ mask, const float* __restrict__ s,
                      const float* __restrict__ c, const float* __restrict__ m,
                      const float* __restrict__ rd, const __bf16* __restrict__ ftT,
                      float* __restrict__ omid) {
  __shared__ __attribute__((aligned(16))) __bf16 sP[16 * 32];
  __shared__ __attribute__((aligned(16))) unsigned sMaskL[16 * MSTRIDE];
  const int t = threadIdx.x;
  const int lane = t & 31;
  const int wave = t >> 5;
  const int i0 = blockIdx.x * 16;
  const int r = lane & 15;
  const int hi = lane >> 4;
  const int u0a = (wave * 2 + 0) * 16;
  const int u0b = (wave * 2 + 1) * 16;

  // ---- async-stage this block's mask slab: 16 rows x 256 words, 1024 x 16B chunks
#pragma unroll
  for (int k = 0; k < 8; ++k) {
    const int ci = k * 128 + t;                          // 0..1023
    const int row = ci >> 6;                             // 0..15
    const int wi = ci & 63;                              // 16B chunk within row
    const unsigned ldsoff =
        (unsigned)(uintptr_t)(&sMaskL[row * MSTRIDE + wi * 4]);
    const unsigned* gsrc = mask + (size_t)(i0 + row) * NW + wi * 4;
    asm volatile("global_load_async_to_lds_b128 %0, %1, off"
                 :: "v"(ldsoff), "v"(gsrc)
                 : "memory");
  }
  asm volatile("s_wait_asynccnt 0" ::: "memory");
  __syncthreads();

  // P-build mapping: 8 threads per row, 4 cols each
  const int pr = t >> 3;
  const int pq = t & 7;
  const float sv = s[i0 + pr];
  v8f acc0 = {}, acc1 = {};
  for (int g = 0; g < NW; ++g) {
    const int j0 = g * 32;
    __syncthreads();                                     // safe to overwrite sP
    const unsigned word = sMaskL[pr * MSTRIDE + g];
#pragma unroll
    for (int e = 0; e < 4; ++e) {
      const int jj = pq * 4 + e;
      const int j = j0 + jj;
      float p = 0.f;
      if ((word >> jj) & 1u)
        p = __expf(lrelu(sv + c[j]) - m[j]) * rd[j];
      sP[pr * 32 + jj] = (__bf16)p;
    }
    __syncthreads();
    // A-frag from LDS
    const int s0 = hi ? 16 : 0;
    const __bf16* ab = sP + r * 32;
    v16bf Af = join16(*(const v8bf*)(ab + s0), *(const v8bf*)(ab + s0 + 8));
    // B-frags from ftT (L2-resident): 16 contiguous j per lane
    const int jb = j0 + (hi ? 16 : 0);
    const __bf16* b0p = ftT + (size_t)(u0a + r) * NN + jb;
    const __bf16* b1p = ftT + (size_t)(u0b + r) * NN + jb;
    v16bf B0 = join16(*(const v8bf*)b0p, *(const v8bf*)(b0p + 8));
    v16bf B1 = join16(*(const v8bf*)b1p, *(const v8bf*)(b1p + 8));
    acc0 = __builtin_amdgcn_wmma_f32_16x16x32_bf16(false, Af, false, B0, (short)0, acc0, false, false);
    acc1 = __builtin_amdgcn_wmma_f32_16x16x32_bf16(false, Af, false, B1, (short)0, acc1, false, false);
  }
#pragma unroll
  for (int rr = 0; rr < 8; ++rr) {
    const int i = i0 + rr + (hi ? 8 : 0);
    omid[(size_t)i * UU + u0a + r] = acc0[rr];
    omid[(size_t)i * UU + u0b + r] = acc1[rr];
  }
}

// -------------------------------------- final dense: out = out_mid @ dense_w + bias

__global__ void k_dense(const float* __restrict__ omid, const __bf16* __restrict__ dwT,
                        const float* __restrict__ bias, float* __restrict__ out) {
  const int lane = threadIdx.x & 31;
  const int wave = threadIdx.x >> 5;
  const int i0 = blockIdx.x * 16;
  const int r = lane & 15;
  const int hi = lane >> 4;
  const int u0a = (wave * 2 + 0) * 16;
  const int u0b = (wave * 2 + 1) * 16;
  v8f acc0 = {}, acc1 = {};
  for (int k0 = 0; k0 < UU; k0 += 32) {
    const int s0 = k0 + (hi ? 16 : 0);
    const float* arow = omid + (size_t)(i0 + r) * UU + s0;
    float4 a0 = *(const float4*)(arow);
    float4 a1 = *(const float4*)(arow + 4);
    float4 a2 = *(const float4*)(arow + 8);
    float4 a3 = *(const float4*)(arow + 12);
    float av[16] = {a0.x, a0.y, a0.z, a0.w, a1.x, a1.y, a1.z, a1.w,
                    a2.x, a2.y, a2.z, a2.w, a3.x, a3.y, a3.z, a3.w};
    v16bf Af;
#pragma unroll
    for (int e = 0; e < 16; ++e) Af[e] = (__bf16)av[e];
    const int kb = k0 + (hi ? 16 : 0);
    const __bf16* b0p = dwT + (size_t)(u0a + r) * UU + kb;
    const __bf16* b1p = dwT + (size_t)(u0b + r) * UU + kb;
    v16bf B0 = join16(*(const v8bf*)b0p, *(const v8bf*)(b0p + 8));
    v16bf B1 = join16(*(const v8bf*)b1p, *(const v8bf*)(b1p + 8));
    acc0 = __builtin_amdgcn_wmma_f32_16x16x32_bf16(false, Af, false, B0, (short)0, acc0, false, false);
    acc1 = __builtin_amdgcn_wmma_f32_16x16x32_bf16(false, Af, false, B1, (short)0, acc1, false, false);
  }
  const float b0 = bias[u0a + r];
  const float b1 = bias[u0b + r];
#pragma unroll
  for (int rr = 0; rr < 8; ++rr) {
    const int i = i0 + rr + (hi ? 8 : 0);
    out[(size_t)i * UU + u0a + r] = acc0[rr] + b0;
    out[(size_t)i * UU + u0b + r] = acc1[rr] + b1;
  }
}

// --------------------------------------------------------------------- host entry

extern "C" void kernel_launch(void* const* d_in, const int* in_sizes, int n_in,
                              void* d_out, int out_size, void* d_ws, size_t ws_size,
                              hipStream_t stream) {
  (void)in_sizes; (void)n_in; (void)out_size; (void)ws_size;
  const float* feat   = (const float*)d_in[0];
  const float* A      = (const float*)d_in[1];
  const float* kern   = (const float*)d_in[2];
  const float* aself  = (const float*)d_in[3];
  const float* across = (const float*)d_in[4];
  const float* dw     = (const float*)d_in[5];
  const float* db     = (const float*)d_in[6];
  float* out = (float*)d_out;

  char* w = (char*)d_ws;
  size_t o = 0;
  auto take = [&](size_t bytes) -> char* {
    char* p = w + o;
    o += (bytes + 255) & ~(size_t)255;
    return p;
  };
  float*    ft    = (float*)take((size_t)NN * UU * 4);
  float*    omid  = (float*)take((size_t)NN * UU * 4);
  __bf16*   ftT   = (__bf16*)take((size_t)UU * NN * 2);
  __bf16*   kbf   = (__bf16*)take((size_t)UU * FF * 2);
  __bf16*   dwT   = (__bf16*)take((size_t)UU * UU * 2);
  unsigned* mask  = (unsigned*)take((size_t)NN * NW * 4);
  float*    dPart = (float*)take((size_t)DCHUNKS * NN * 4);
  float*    sArr  = (float*)take((size_t)NN * 4);
  float*    cArr  = (float*)take((size_t)NN * 4);
  float*    smax  = (float*)take((size_t)NN * 4);
  float*    mArr  = (float*)take((size_t)NN * 4);
  float*    rdArr = (float*)take((size_t)NN * 4);

  k_cvt_kernel<<<(UU * FF) / 256, 256, 0, stream>>>(kern, kbf);
  k_cvt_dw<<<(UU * UU) / 256, 256, 0, stream>>>(dw, dwT);
  k_init_smax<<<NN / 256, 256, 0, stream>>>(smax);
  k_ft<<<NN / 16, 128, 0, stream>>>(feat, kbf, ft);
  k_tr<<<dim3(UU / 32, NN / 32), dim3(32, 8), 0, stream>>>(ft, ftT);
  k_sc<<<NN, UU, 0, stream>>>(ft, aself, across, sArr, cArr);
  k_mask<<<dim3(NW, 32), 256, 0, stream>>>(A, sArr, mask, smax);
  k_colmax<<<NN / 256, 256, 0, stream>>>(smax, cArr, mArr);
  k_dpart<<<dim3(NW, DCHUNKS / 8), 256, 0, stream>>>(mask, sArr, cArr, mArr, dPart);
  k_dred<<<NN / 256, 256, 0, stream>>>(dPart, rdArr);
  k_att<<<NN / 16, 128, 0, stream>>>(mask, sArr, cArr, mArr, rdArr, ftT, omid);
  k_dense<<<NN / 16, 128, 0, stream>>>(omid, dwT, db, out);
}
